// FeatureAlign_V2_84988812853772
// MI455X (gfx1250) — compile-verified, matrix operand order checked by reading, not run on hardware
//
#include <hip/hip_runtime.h>
#include <hip/hip_bf16.h>

// ---------------- problem constants ----------------
#define BB 4
#define CC 256          // C
#define HH 64
#define WW 64
#define HS 32           // H/2
#define WS 32           // W/2
#define COUT 256
#define KTAPS 9
#define KTOT (KTAPS*CC) // 2304
#define NKB  (KTOT/32)  // 72 K-blocks of 32
#define NPIX (BB*HH*WW) // 16384
#define HWP  (HH*WW)    // 4096

typedef __attribute__((ext_vector_type(16))) __bf16        v16bf;
typedef __attribute__((ext_vector_type(8)))  float         v8f;
typedef __attribute__((ext_vector_type(4)))  unsigned int  v4u;

union Frag {
    v4u          q[2];
    unsigned int u[8];
    v16bf        bf;
};

__device__ __forceinline__ unsigned int pack_bf16x2(float a, float b) {
    unsigned int ua = __float_as_uint(a);
    unsigned int ub = __float_as_uint(b);
    ua = (ua + 0x7FFFu + ((ua >> 16) & 1u)) >> 16;      // RNE float->bf16
    ub = (ub + 0x7FFFu + ((ub >> 16) & 1u)) >> 16;
    return ua | (ub << 16);
}

__device__ __forceinline__ int iclamp(int v, int lo, int hi) {
    return v < lo ? lo : (v > hi ? hi : v);
}

// ---------------------------------------------------------------------------
// Kernel 1: bilinear 2x upsample, NCHW feat_s -> NHWC feat_up (unscaled)
// grid: B * H * (C/64) blocks, 256 threads
// ---------------------------------------------------------------------------
__global__ void k_upsample(const float* __restrict__ feat_s,
                           float* __restrict__ feat_up /* [B][H][W][C] */) {
    __shared__ float ls[2 * 64 * 33];   // [y(2)][c(64)][x(32)+pad]

    const int bid = blockIdx.x;
    const int cb  = bid & 3;            // channel tile of 64
    const int h   = (bid >> 2) & 63;
    const int b   = bid >> 8;
    const int t   = threadIdx.x;

    const float sy  = h * 0.5f - 0.25f;
    const float y0f = floorf(sy);
    const float ly  = sy - y0f;
    const int iy0 = iclamp((int)y0f, 0, HS - 1);
    const int iy1 = iclamp((int)y0f + 1, 0, HS - 1);

    #pragma unroll
    for (int i = 0; i < 16; ++i) {
        int j  = t + i * 256;           // 0..4095
        int x  = j & 31;
        int y  = (j >> 5) & 1;
        int cl = j >> 6;                // 0..63
        int ys = y ? iy1 : iy0;
        ls[(y * 64 + cl) * 33 + x] =
            feat_s[(((size_t)b * CC + cb * 64 + cl) * HS + ys) * WS + x];
    }
    __syncthreads();

    const int c_off = t & 63;
    const int wg    = t >> 6;           // 0..3
    #pragma unroll
    for (int i = 0; i < 16; ++i) {
        int w = wg * 16 + i;
        float sx  = w * 0.5f - 0.25f;
        float x0f = floorf(sx);
        float lx  = sx - x0f;
        int ix0 = iclamp((int)x0f, 0, WS - 1);
        int ix1 = iclamp((int)x0f + 1, 0, WS - 1);
        float v0 = (1.0f - lx) * ls[(0 * 64 + c_off) * 33 + ix0] + lx * ls[(0 * 64 + c_off) * 33 + ix1];
        float v1 = (1.0f - lx) * ls[(1 * 64 + c_off) * 33 + ix0] + lx * ls[(1 * 64 + c_off) * 33 + ix1];
        float val = (1.0f - ly) * v0 + ly * v1;
        feat_up[(((size_t)b * HH + h) * WW + w) * CC + cb * 64 + c_off] = val;
    }
}

// ---------------------------------------------------------------------------
// Kernel 2: 1x1 offset conv (18 x 768) + ReLU -> offs NHWC [B][H][W][18]
// grid: NPIX/256 blocks, 256 threads (one pixel per lane)
// ---------------------------------------------------------------------------
__global__ void k_offset(const float* __restrict__ feat_l,   // [B][2C][H][W]
                         const float* __restrict__ feat_up,  // [B][H][W][C] NHWC
                         const float* __restrict__ w_off,    // [18][3C]
                         float* __restrict__ offs) {         // [B][H][W][18]
    __shared__ float lch[256 * 33];     // 256 pixels x 32 channels (+pad)

    const int t  = threadIdx.x;
    const int n0 = blockIdx.x * 256;
    const int n  = n0 + t;
    const int b  = n >> 12;
    const int hw = n & (HWP - 1);

    float acc[18];
    #pragma unroll
    for (int o = 0; o < 18; ++o) acc[o] = 0.0f;

    // part 1: feat_l channels 0..511 (coalesced: lanes = consecutive pixels)
    for (int c = 0; c < 2 * CC; ++c) {
        float v = feat_l[((size_t)b * (2 * CC) + c) * HWP + hw];
        #pragma unroll
        for (int o = 0; o < 18; ++o)
            acc[o] += w_off[o * (3 * CC) + c] * v;
    }

    // part 2: 2*feat_up channels, staged through LDS (coalesced NHWC reads)
    for (int cbk = 0; cbk < 8; ++cbk) {
        __syncthreads();
        #pragma unroll
        for (int i = 0; i < 32; ++i) {
            int pp = (t >> 5) + i * 8;                  // 0..255
            int cl = t & 31;
            lch[pp * 33 + cl] =
                2.0f * feat_up[(size_t)(n0 + pp) * CC + cbk * 32 + cl];
        }
        __syncthreads();
        #pragma unroll
        for (int cl = 0; cl < 32; ++cl) {
            float v = lch[t * 33 + cl];
            int c = 2 * CC + cbk * 32 + cl;
            #pragma unroll
            for (int o = 0; o < 18; ++o)
                acc[o] += w_off[o * (3 * CC) + c] * v;
        }
    }

    #pragma unroll
    for (int o = 0; o < 18; ++o)
        offs[(size_t)n * 18 + o] = fmaxf(acc[o], 0.0f);
}

// ---------------------------------------------------------------------------
// Kernel 3: pre-swizzle w_dcn into bf16 WMMA A-fragment layout.
// K order: Kg = tap*256 + c.  Layout: [mtile(16)][kb(72)][lane(32)][dword(8)]
// A 16-bit layout (ISA 7.12.2): lane&15 = M, half=lane>>4,
//   K(v,sub,half) = (v>=4?16:0) + (half<<3) + ((v&3)<<1) + sub
// grid: 16*72 blocks, 256 threads
// ---------------------------------------------------------------------------
__global__ void k_pack_w(const float* __restrict__ w_dcn,    // [Cout][C][9]
                         unsigned int* __restrict__ apack) {
    const int bid = blockIdx.x;
    const int tm  = bid / NKB;          // m-tile 0..15
    const int kb  = bid % NKB;          // k-block 0..71
    const int u    = threadIdx.x;
    const int lane = u & 31;
    const int v    = u >> 5;            // 0..7
    const int half = lane >> 4;
    const int m    = lane & 15;

    float f[2];
    #pragma unroll
    for (int sub = 0; sub < 2; ++sub) {
        int kk  = ((v >= 4) ? 16 : 0) + (half << 3) + ((v & 3) << 1) + sub;
        int Kg  = kb * 32 + kk;
        int tap = Kg >> 8;
        int c   = Kg & 255;
        f[sub]  = w_dcn[((size_t)(tm * 16 + m) * CC + c) * KTAPS + tap];
    }
    apack[((size_t)(tm * NKB + kb) * 32 + lane) * 8 + v] = pack_bf16x2(f[0], f[1]);
}

// ---------------------------------------------------------------------------
// Kernel 4: deformable 3x3 conv as bf16 WMMA GEMM.
// Block = 32 pixels (two 16-wide N tiles) x all 256 Cout; 256 threads, 8 waves.
// Wave wv owns M-tiles {2wv, 2wv+1} x N-tiles {0,1}: 4 WMMAs / K-block with
// full A/B register reuse. B fragments double-buffered in LDS: one barrier
// per K-block; next K-block's bilinear gather overlaps current WMMAs.
// Gather uses uniform scalar base + 32-bit byte offsets -> saddr-form loads.
// grid: NPIX/32 blocks
// ---------------------------------------------------------------------------
__global__ void k_deform_wmma(const float* __restrict__ feat_up,      // NHWC
                              const float* __restrict__ offs,         // [n][18]
                              const unsigned int* __restrict__ apack,
                              const float* __restrict__ b_dcn,
                              float* __restrict__ out) {              // [B][Cout][H][W]
    __shared__ float4       swt[KTAPS][32];      // masked bilinear corner weights
    __shared__ int4         sad[KTAPS][32];      // corner BYTE offsets within batch
    __shared__ unsigned int bfrag[2][2][32 * 8]; // [buf][ntile][lane*8+v]

    const int t    = threadIdx.x;
    const int lane = t & 31;
    const int wv   = t >> 5;            // wave id 0..7
    const int n0   = blockIdx.x * 32;
    const int b    = n0 >> 12;
    const int hw0  = n0 & (HWP - 1);
    const int h    = hw0 >> 6;
    const int w0   = hw0 & 63;          // 0 or 32
    // wave-uniform batch base pointer (kept in SGPRs -> saddr-form gathers)
    const char* fb = (const char*)feat_up + (size_t)b * (HH * WW * CC) * sizeof(float);

    // ---- per (tap, pixel) sampling setup: 9 taps x 32 pixels ----
    for (int idx = t; idx < KTAPS * 32; idx += 256) {
        const int p   = idx & 31;
        const int tap = idx >> 5;
        const float dy = offs[(size_t)(n0 + p) * 18 + tap * 2 + 0];
        const float dx = offs[(size_t)(n0 + p) * 18 + tap * 2 + 1];
        const float py = (float)h + (float)(tap / 3 - 1) + dy;
        const float px = (float)(w0 + p) + (float)(tap % 3 - 1) + dx;
        const float y0f = floorf(py), x0f = floorf(px);
        const float ly = py - y0f,    lx = px - x0f;
        const int iy0 = (int)y0f, ix0 = (int)x0f;
        const float cw[4] = { (1.f - ly) * (1.f - lx), (1.f - ly) * lx,
                              ly * (1.f - lx),         ly * lx };
        const int cy[4] = { iy0, iy0, iy0 + 1, iy0 + 1 };
        const int cx[4] = { ix0, ix0 + 1, ix0, ix0 + 1 };
        float4 sv;
        int4   av;
        float* svp = &sv.x;
        int*   avp = &av.x;
        #pragma unroll
        for (int j = 0; j < 4; ++j) {
            bool valid = (cy[j] >= 0) & (cy[j] < HH) & (cx[j] >= 0) & (cx[j] < WW);
            svp[j] = valid ? cw[j] : 0.0f;
            avp[j] = (iclamp(cy[j], 0, HH - 1) * WW + iclamp(cx[j], 0, WW - 1))
                     * (CC * (int)sizeof(float));
        }
        swt[tap][p] = sv;
        sad[tap][p] = av;
    }

    // cooperative bilinear gather of one K-block into B fragments (both N tiles)
    const int gp    = t >> 4;           // pixel low 0..15
    const int cpair = (t & 15) * 2;     // even channel within K-block
    const int halfb = (cpair >> 3) & 1;
    const int vb_i  = ((cpair >= 16) ? 4 : 0) + ((cpair & 7) >> 1);
    const int laneb = gp + (halfb << 4);

    auto buildB = [&](int kbn, int buf) {
        const int tap = kbn >> 3;
        const unsigned cbyte = (unsigned)(((kbn & 7) * 32 + cpair) * sizeof(float));
        #pragma unroll
        for (int nt = 0; nt < 2; ++nt) {
            const int pp = gp + nt * 16;
            const int4   a = sad[tap][pp];
            const float4 s = swt[tap][pp];
            const float2 q0 = *(const float2*)(fb + ((unsigned)a.x + cbyte));
            const float2 q1 = *(const float2*)(fb + ((unsigned)a.y + cbyte));
            const float2 q2 = *(const float2*)(fb + ((unsigned)a.z + cbyte));
            const float2 q3 = *(const float2*)(fb + ((unsigned)a.w + cbyte));
            const float va = s.x * q0.x + s.y * q1.x + s.z * q2.x + s.w * q3.x;
            const float vb = s.x * q0.y + s.y * q1.y + s.z * q2.y + s.w * q3.y;
            bfrag[buf][nt][laneb * 8 + vb_i] = pack_bf16x2(va, vb);
        }
    };

    v8f acc00 = {}, acc01 = {}, acc10 = {}, acc11 = {};
    const int tm0 = wv * 2, tm1 = wv * 2 + 1;

    __syncthreads();        // setup tables ready
    buildB(0, 0);
    __syncthreads();

    for (int kb = 0; kb < NKB; ++kb) {
        const int buf = kb & 1;

        // load current B fragments from LDS (2x ds_load_b128 each)
        Frag fb0, fb1;
        {
            const v4u* bp0 = (const v4u*)&bfrag[buf][0][lane * 8];
            const v4u* bp1 = (const v4u*)&bfrag[buf][1][lane * 8];
            fb0.q[0] = bp0[0]; fb0.q[1] = bp0[1];
            fb1.q[0] = bp1[0]; fb1.q[1] = bp1[1];
        }
        // load A fragments (2x global_load_b128 each, from packed weights)
        Frag fa0, fa1;
        {
            const v4u* ap0 = (const v4u*)(apack + ((size_t)(tm0 * NKB + kb) * 32 + lane) * 8);
            const v4u* ap1 = (const v4u*)(apack + ((size_t)(tm1 * NKB + kb) * 32 + lane) * 8);
            fa0.q[0] = ap0[0]; fa0.q[1] = ap0[1];
            fa1.q[0] = ap1[0]; fa1.q[1] = ap1[1];
        }

        // start gathering the next K-block while WMMAs run
        if (kb + 1 < NKB) buildB(kb + 1, buf ^ 1);

        acc00 = __builtin_amdgcn_wmma_f32_16x16x32_bf16(
            false, fa0.bf, false, fb0.bf, (short)0, acc00, false, false);
        acc01 = __builtin_amdgcn_wmma_f32_16x16x32_bf16(
            false, fa0.bf, false, fb1.bf, (short)0, acc01, false, false);
        acc10 = __builtin_amdgcn_wmma_f32_16x16x32_bf16(
            false, fa1.bf, false, fb0.bf, (short)0, acc10, false, false);
        acc11 = __builtin_amdgcn_wmma_f32_16x16x32_bf16(
            false, fa1.bf, false, fb1.bf, (short)0, acc11, false, false);

        __syncthreads();    // next buffer fully built / current fully consumed
    }

    // ---- bias + ReLU + NCHW store (4 output tiles) ----
    const int nl = lane & 15;
    const int mh = (lane >> 4) * 8;
    #pragma unroll
    for (int r = 0; r < 8; ++r) {
        const int m0 = tm0 * 16 + r + mh;
        const int m1 = tm1 * 16 + r + mh;
        const float bi0 = b_dcn[m0];
        const float bi1 = b_dcn[m1];
        out[((size_t)b * COUT + m0) * HWP + hw0 + nl]      = fmaxf(acc00[r] + bi0, 0.0f);
        out[((size_t)b * COUT + m0) * HWP + hw0 + 16 + nl] = fmaxf(acc01[r] + bi0, 0.0f);
        out[((size_t)b * COUT + m1) * HWP + hw0 + nl]      = fmaxf(acc10[r] + bi1, 0.0f);
        out[((size_t)b * COUT + m1) * HWP + hw0 + 16 + nl] = fmaxf(acc11[r] + bi1, 0.0f);
    }
}

// ---------------------------------------------------------------------------
extern "C" void kernel_launch(void* const* d_in, const int* in_sizes, int n_in,
                              void* d_out, int out_size, void* d_ws, size_t ws_size,
                              hipStream_t stream) {
    const float* feat_l  = (const float*)d_in[0];   // [4][512][64][64]
    const float* feat_s  = (const float*)d_in[1];   // [4][256][32][32]
    const float* w_off   = (const float*)d_in[2];   // [18][768]
    const float* w_dcn   = (const float*)d_in[3];   // [256][256][3][3]
    const float* b_dcn   = (const float*)d_in[4];   // [256]
    float*       out     = (float*)d_out;           // [4][256][64][64]

    float* ws = (float*)d_ws;
    float*        feat_up = ws;                                  // 4,194,304 f
    float*        offsbuf = ws + (size_t)BB * HH * WW * CC;      // 294,912 f
    unsigned int* apack   = (unsigned int*)(offsbuf + (size_t)NPIX * 18); // 294,912 dw

    k_upsample<<<dim3(BB * HH * (CC / 64)), dim3(256), 0, stream>>>(feat_s, feat_up);
    k_offset  <<<dim3(NPIX / 256),          dim3(256), 0, stream>>>(feat_l, feat_up, w_off, offsbuf);
    k_pack_w  <<<dim3(16 * NKB),            dim3(256), 0, stream>>>(w_dcn, apack);
    k_deform_wmma<<<dim3(NPIX / 32),        dim3(256), 0, stream>>>(feat_up, offsbuf, apack,
                                                                    b_dcn, out);
}
